// PipelinedMoEBlock_84301618086178
// MI455X (gfx1250) — matrix operand order repaired and checked by reading, not compiled, not run on hardware
//
#include <hip/hip_runtime.h>
#include <hip/hip_bf16.h>

typedef _Float16 h16;
typedef __attribute__((ext_vector_type(4)))  _Float16 v4h;
typedef __attribute__((ext_vector_type(8)))  _Float16 v8h;
typedef __attribute__((ext_vector_type(16))) _Float16 v16h;
typedef __attribute__((ext_vector_type(8)))  float    v8f;

#define DIM 768
#define FF  3072
#define NE  8
#define NH  12
#define HD  64
#define NTOK 4096
#define TSEQ 1024

__device__ __forceinline__ v16h cat8(v8h lo, v8h hi) {
  return __builtin_shufflevector(lo, hi, 0,1,2,3,4,5,6,7,8,9,10,11,12,13,14,15);
}

__device__ __forceinline__ v8f wmma16(v16h a, v16h b, v8f c) {
  return __builtin_amdgcn_wmma_f32_16x16x32_f16(false, a, false, b, (short)0, c, false, false);
}

// ---- load an A fragment (row-major f16, contiguous k = 16h + s order, same as B) ----
__device__ __forceinline__ v16h loadA(const h16* __restrict__ arow, int k0, int hi8) {
  v8h lo = *(const v8h*)(arow + k0 + 2 * hi8);        // K = k0 + 2*base .. +7
  v8h hi = *(const v8h*)(arow + k0 + 2 * hi8 + 8);   // K = k0 + 2*base+8 .. +15
  return cat8(lo, hi);
}

// ---- software-pipelined mainloop: one wave -> 32x64 tile (8 accumulators) ----
// Fragments prefetched one k-step ahead so WMMAs overlap the loads.
// Callers must be __launch_bounds__(128, 1) so ~180 VGPRs fit without spills.
__device__ __forceinline__ void gemm_mainloop2(const h16* __restrict__ A, int lda,
                                               const h16* __restrict__ Bt, int ldb,
                                               int m0, int n0, int K, v8f acc[8]) {
  const int lane = threadIdx.x & 31;
  const int lrow = lane & 15;
  const int hi8  = (lane >> 4) << 3;        // 0 or 8
  const int hi16 = hi8 << 1;                // 0 or 16
  const h16* ar0 = A + (size_t)(m0 + lrow) * lda;
  const h16* ar1 = ar0 + (size_t)16 * lda;
  const h16* bp0 = Bt + (size_t)(n0 + lrow) * ldb + hi16;
  const h16* bp1 = bp0 + (size_t)16 * ldb;
  const h16* bp2 = bp0 + (size_t)32 * ldb;
  const h16* bp3 = bp0 + (size_t)48 * ldb;

  v16h a0 = loadA(ar0, 0, hi8);
  v16h a1 = loadA(ar1, 0, hi8);
  v16h b0 = *(const v16h*)(bp0);
  v16h b1 = *(const v16h*)(bp1);
  v16h b2 = *(const v16h*)(bp2);
  v16h b3 = *(const v16h*)(bp3);

  int k0 = 0;
  for (; k0 + 32 < K; k0 += 32) {
    const int kn = k0 + 32;
    // issue next k-step's fragment loads first (latency hiding)
    v16h na0 = loadA(ar0, kn, hi8);
    v16h na1 = loadA(ar1, kn, hi8);
    v16h nb0 = *(const v16h*)(bp0 + kn);
    v16h nb1 = *(const v16h*)(bp1 + kn);
    v16h nb2 = *(const v16h*)(bp2 + kn);
    v16h nb3 = *(const v16h*)(bp3 + kn);
    __builtin_prefetch(bp0 + kn + 256, 0, 1);
    __builtin_prefetch(bp2 + kn + 256, 0, 1);
    acc[0] = wmma16(a0, b0, acc[0]);
    acc[1] = wmma16(a0, b1, acc[1]);
    acc[2] = wmma16(a0, b2, acc[2]);
    acc[3] = wmma16(a0, b3, acc[3]);
    acc[4] = wmma16(a1, b0, acc[4]);
    acc[5] = wmma16(a1, b1, acc[5]);
    acc[6] = wmma16(a1, b2, acc[6]);
    acc[7] = wmma16(a1, b3, acc[7]);
    a0 = na0; a1 = na1;
    b0 = nb0; b1 = nb1; b2 = nb2; b3 = nb3;
  }
  acc[0] = wmma16(a0, b0, acc[0]);
  acc[1] = wmma16(a0, b1, acc[1]);
  acc[2] = wmma16(a0, b2, acc[2]);
  acc[3] = wmma16(a0, b3, acc[3]);
  acc[4] = wmma16(a1, b0, acc[4]);
  acc[5] = wmma16(a1, b1, acc[5]);
  acc[6] = wmma16(a1, b2, acc[6]);
  acc[7] = wmma16(a1, b3, acc[7]);
}

// ---- weight transpose + f16 cast: Wt[n*K+k] = (h16)W[k*N+n] ----
__global__ void cvtT_kernel(const float* __restrict__ W, h16* __restrict__ Wt,
                            int K, int N, long inStride, long outStride) {
  const float* Wp = W + (long)blockIdx.z * inStride;
  h16* Wtp = Wt + (long)blockIdx.z * outStride;
  long id = (long)blockIdx.x * blockDim.x + threadIdx.x;
  long total = (long)K * N;
  if (id >= total) return;
  int n = (int)(id / K), kk = (int)(id % K);
  Wtp[(size_t)n * K + kk] = (h16)Wp[(size_t)kk * N + n];
}

// ---- LayerNorm (row per block, LDS tree reduce); optional f32 output ----
__global__ void ln_kernel(const float* __restrict__ x, const float* __restrict__ g,
                          const float* __restrict__ b, float* __restrict__ outf,
                          h16* __restrict__ outh) {
  int row = blockIdx.x, tid = threadIdx.x;
  const float* xr = x + (size_t)row * DIM;
  float v0 = xr[tid], v1 = xr[tid + 256], v2 = xr[tid + 512];
  __shared__ float r1[256], r2[256];
  r1[tid] = v0 + v1 + v2;
  r2[tid] = v0 * v0 + v1 * v1 + v2 * v2;
  __syncthreads();
  for (int off = 128; off > 0; off >>= 1) {
    if (tid < off) { r1[tid] += r1[tid + off]; r2[tid] += r2[tid + off]; }
    __syncthreads();
  }
  float mean = r1[0] * (1.0f / DIM);
  float var  = r2[0] * (1.0f / DIM) - mean * mean;
  float inv  = rsqrtf(var + 1e-5f);
#pragma unroll
  for (int j = 0; j < 3; ++j) {
    int c = tid + j * 256;
    float y = (xr[c] - mean) * inv * g[c] + b[c];
    if (outf) outf[(size_t)row * DIM + c] = y;
    outh[(size_t)row * DIM + c] = (h16)y;
  }
}

// ---- QKV GEMM: A(4096x768) @ WqkvT -> q,k row-major f16 ; v transposed ----
__global__ void __launch_bounds__(128, 1)
gemm_qkv(const h16* __restrict__ A, const h16* __restrict__ Bt,
         h16* __restrict__ q, h16* __restrict__ k, h16* __restrict__ vT) {
  int wave = threadIdx.x >> 5;
  int m0 = (blockIdx.x * 8 + wave * 2) * 16;
  int n0 = blockIdx.y * 64;
  v8f z = {};
  v8f acc[8] = {z, z, z, z, z, z, z, z};
  gemm_mainloop2(A, DIM, Bt, DIM, m0, n0, DIM, acc);
  int lane = threadIdx.x & 31, lrow = lane & 15, hi8 = (lane >> 4) << 3;
#pragma unroll
  for (int mt = 0; mt < 2; ++mt)
#pragma unroll
    for (int nt = 0; nt < 4; ++nt)
#pragma unroll
      for (int r = 0; r < 8; ++r) {
        int nglob = n0 + nt * 16 + lrow;
        int row = m0 + mt * 16 + r + hi8;
        h16 hv = (h16)acc[mt * 4 + nt][r];
        if (nglob < DIM) {
          q[(size_t)row * DIM + nglob] = hv;
        } else if (nglob < 2 * DIM) {
          k[(size_t)row * DIM + (nglob - DIM)] = hv;
        } else {
          int c = nglob - 2 * DIM;
          int hh = c >> 6, d = c & 63;
          int bb = row >> 10, t = row & 1023;
          vT[((size_t)((bb * NH + hh) * HD + d)) * TSEQ + t] = hv;
        }
      }
}

// ---- Flash attention: one wave = one 16-row q tile of one (b,h) ----
__global__ void __launch_bounds__(128, 1)
attn_kernel(const h16* __restrict__ q, const h16* __restrict__ k,
            const h16* __restrict__ vT, h16* __restrict__ o) {
  const int bh = blockIdx.x, b = bh / NH, h = bh % NH;
  const int wave = threadIdx.x >> 5;
  const int qt = blockIdx.y * 4 + wave;
  const int lane = threadIdx.x & 31;
  const int lrow = lane & 15;
  const int hi8  = (lane >> 4) << 3;
  const int hi16 = hi8 << 1;

  __shared__ __align__(16) h16 pbuf[4][16][32];

  const h16* qrow = q + (size_t)(b * TSEQ + qt * 16 + lrow) * DIM + h * HD;
  v16h aq0 = loadA(qrow, 0, hi8);
  v16h aq1 = loadA(qrow, 32, hi8);

  float m[8], l[8];
  v8f z = {};
  v8f acc[4] = {z, z, z, z};
#pragma unroll
  for (int r = 0; r < 8; ++r) { m[r] = -1e30f; l[r] = 0.f; }

  for (int kv0 = 0; kv0 < TSEQ; kv0 += 32) {
    const h16* kb = k + (size_t)(b * TSEQ + kv0 + lrow) * DIM + h * HD;
    v8f s0 = z, s1 = z;
    s0 = wmma16(aq0, *(const v16h*)(kb + hi16), s0);
    s0 = wmma16(aq1, *(const v16h*)(kb + 32 + hi16), s0);
    const h16* kb1 = kb + 16 * DIM;
    s1 = wmma16(aq0, *(const v16h*)(kb1 + hi16), s1);
    s1 = wmma16(aq1, *(const v16h*)(kb1 + 32 + hi16), s1);

    float p0[8], p1[8], sc[8];
#pragma unroll
    for (int r = 0; r < 8; ++r) {
      float a0 = s0[r] * 0.125f, a1 = s1[r] * 0.125f;
      float mx = fmaxf(a0, a1);
#pragma unroll
      for (int d = 1; d < 16; d <<= 1) mx = fmaxf(mx, __shfl_xor(mx, d, 32));
      float mn = fmaxf(m[r], mx);
      sc[r] = __expf(m[r] - mn);
      p0[r] = __expf(a0 - mn);
      p1[r] = __expf(a1 - mn);
      float rs = p0[r] + p1[r];
#pragma unroll
      for (int d = 1; d < 16; d <<= 1) rs += __shfl_xor(rs, d, 32);
      l[r] = l[r] * sc[r] + rs;
      m[r] = mn;
    }
#pragma unroll
    for (int nt = 0; nt < 4; ++nt)
#pragma unroll
      for (int r = 0; r < 8; ++r) acc[nt][r] *= sc[r];

    // C-layout -> A-layout via LDS
#pragma unroll
    for (int r = 0; r < 8; ++r) {
      pbuf[wave][r + hi8][lrow] = (h16)p0[r];
      pbuf[wave][r + hi8][16 + lrow] = (h16)p1[r];
    }
    __syncthreads();
    const h16* pr = &pbuf[wave][lrow][0];
    v16h pa = cat8(*(const v8h*)(pr + 2 * hi8), *(const v8h*)(pr + 2 * hi8 + 8));
    __syncthreads();

    const h16* vb = vT + (size_t)(bh * HD + lrow) * TSEQ + kv0 + hi16;
#pragma unroll
    for (int nt = 0; nt < 4; ++nt) {
      v16h bv = *(const v16h*)(vb + (size_t)(nt * 16) * TSEQ);
      acc[nt] = wmma16(pa, bv, acc[nt]);
    }
  }
#pragma unroll
  for (int nt = 0; nt < 4; ++nt)
#pragma unroll
    for (int r = 0; r < 8; ++r) {
      float ov = acc[nt][r] / l[r];
      o[(size_t)(b * TSEQ + qt * 16 + r + hi8) * DIM + h * HD + nt * 16 + lrow] = (h16)ov;
    }
}

// ---- Proj GEMM fused with residual: out = x + attn @ Wproj ----
__global__ void __launch_bounds__(128, 1)
gemm_proj(const h16* __restrict__ A, const h16* __restrict__ Bt,
          const float* __restrict__ x, float* __restrict__ out) {
  int wave = threadIdx.x >> 5;
  int m0 = (blockIdx.x * 8 + wave * 2) * 16;
  int n0 = blockIdx.y * 64;
  v8f z = {};
  v8f acc[8] = {z, z, z, z, z, z, z, z};
  gemm_mainloop2(A, DIM, Bt, DIM, m0, n0, DIM, acc);
  int lane = threadIdx.x & 31, lrow = lane & 15, hi8 = (lane >> 4) << 3;
#pragma unroll
  for (int mt = 0; mt < 2; ++mt)
#pragma unroll
    for (int nt = 0; nt < 4; ++nt)
#pragma unroll
      for (int r = 0; r < 8; ++r) {
        size_t idx = (size_t)(m0 + mt * 16 + r + hi8) * DIM + n0 + nt * 16 + lrow;
        out[idx] = x[idx] + acc[mt * 4 + nt][r];
      }
}

// ---- Router logits (fp32) ----
__global__ void router_kernel(const float* __restrict__ xf, const float* __restrict__ Wr,
                              float* __restrict__ logits) {
  int id = blockIdx.x * blockDim.x + threadIdx.x;
  if (id >= NTOK * NE) return;
  int t = id >> 3, e = id & 7;
  const float* xr = xf + (size_t)t * DIM;
  float s = 0.f;
  for (int d = 0; d < DIM; ++d) s += xr[d] * Wr[d * NE + e];
  logits[id] = s;
}

__global__ void zero_cnt(int* cnt) { if (threadIdx.x < NE) cnt[threadIdx.x] = 0; }

// ---- Top-2 + softmax weights + expert list compaction ----
__global__ void topk_kernel(const float* __restrict__ logits, int* __restrict__ cnt,
                            int* __restrict__ list, float* __restrict__ wgt,
                            int* __restrict__ invmap) {
  int t = blockIdx.x * blockDim.x + threadIdx.x;
  if (t >= NTOK) return;
  const float* L = logits + (size_t)t * NE;
  int i0 = 0; float v0 = L[0];
  for (int e = 1; e < NE; ++e) { float v = L[e]; if (v > v0) { v0 = v; i0 = e; } }
  int i1 = (i0 == 0) ? 1 : 0; float v1 = L[i1];
  for (int e = 0; e < NE; ++e) {
    if (e == i0) continue;
    float v = L[e]; if (v > v1) { v1 = v; i1 = e; }
  }
  float e1 = __expf(v1 - v0);
  float w0 = 1.f / (1.f + e1), w1 = e1 / (1.f + e1);
  int p0 = atomicAdd(&cnt[i0], 1);
  list[i0 * NTOK + p0] = t; wgt[i0 * NTOK + p0] = w0; invmap[i0 * NTOK + p0] = t * 2;
  int p1 = atomicAdd(&cnt[i1], 1);
  list[i1 * NTOK + p1] = t; wgt[i1 * NTOK + p1] = w1; invmap[i1 * NTOK + p1] = t * 2 + 1;
}

// ---- Gather expert rows (zero-pad to 16-row tiles) ----
__global__ void gather_kernel(const h16* __restrict__ xh, const int* __restrict__ list,
                              const int* __restrict__ cntp, h16* __restrict__ Ag) {
  int row = blockIdx.x, tid = threadIdx.x;    // block = 192, 4 h16 each
  int cnt = *cntp;
  int rc = (cnt + 15) & ~15;
  if (row >= rc) return;
  v4h* dst = (v4h*)(Ag + (size_t)row * DIM + tid * 4);
  if (row < cnt) {
    *dst = *(const v4h*)(xh + (size_t)list[row] * DIM + tid * 4);
  } else {
    v4h zz = {};
    *dst = zz;
  }
}

// ---- FFN1: h = gelu(Ag @ W1e) as f16 ----
__global__ void __launch_bounds__(128, 1)
ffn1_kernel(const h16* __restrict__ Ag, const h16* __restrict__ W1t,
            const int* __restrict__ cntp, h16* __restrict__ H) {
  int wave = threadIdx.x >> 5;
  int m0 = (blockIdx.x * 8 + wave * 2) * 16;
  if (m0 >= *cntp) return;
  int n0 = blockIdx.y * 64;
  v8f z = {};
  v8f acc[8] = {z, z, z, z, z, z, z, z};
  gemm_mainloop2(Ag, DIM, W1t, DIM, m0, n0, DIM, acc);
  int lane = threadIdx.x & 31, lrow = lane & 15, hi8 = (lane >> 4) << 3;
#pragma unroll
  for (int mt = 0; mt < 2; ++mt)
#pragma unroll
    for (int nt = 0; nt < 4; ++nt)
#pragma unroll
      for (int r = 0; r < 8; ++r) {
        float v = acc[mt * 4 + nt][r];
        float gl = 0.5f * v * (1.0f + erff(v * 0.70710678118f));
        H[(size_t)(m0 + mt * 16 + r + hi8) * FF + n0 + nt * 16 + lrow] = (h16)gl;
      }
}

// ---- FFN2: weighted rows scattered to per-token slot buffer (deterministic) ----
__global__ void __launch_bounds__(128, 1)
ffn2_kernel(const h16* __restrict__ H, const h16* __restrict__ W2t,
            const int* __restrict__ cntp, const float* __restrict__ wgt,
            const int* __restrict__ invmap, float* __restrict__ ybuf) {
  int wave = threadIdx.x >> 5;
  int m0 = (blockIdx.x * 8 + wave * 2) * 16;
  int cnt = *cntp;
  if (m0 >= cnt) return;
  int n0 = blockIdx.y * 64;
  v8f z = {};
  v8f acc[8] = {z, z, z, z, z, z, z, z};
  gemm_mainloop2(H, FF, W2t, FF, m0, n0, FF, acc);
  int lane = threadIdx.x & 31, lrow = lane & 15, hi8 = (lane >> 4) << 3;
#pragma unroll
  for (int mt = 0; mt < 2; ++mt)
#pragma unroll
    for (int nt = 0; nt < 4; ++nt)
#pragma unroll
      for (int r = 0; r < 8; ++r) {
        int i = m0 + mt * 16 + r + hi8;
        if (i < cnt) {
          int slot = invmap[i];
          ybuf[(size_t)slot * DIM + n0 + nt * 16 + lrow] = wgt[i] * acc[mt * 4 + nt][r];
        }
      }
}

// ---- Final fixed-order sum: out += y[slot0] + y[slot1] ----
__global__ void final_kernel(const float* __restrict__ ybuf, float* __restrict__ out) {
  long id = (long)blockIdx.x * blockDim.x + threadIdx.x;
  if (id >= (long)NTOK * DIM) return;
  long t = id / DIM, n = id % DIM;
  out[id] += ybuf[(size_t)(t * 2) * DIM + n] + ybuf[(size_t)(t * 2 + 1) * DIM + n];
}

static inline size_t alignup(size_t x) { return (x + 255) & ~(size_t)255; }

extern "C" void kernel_launch(void* const* d_in, const int* in_sizes, int n_in,
                              void* d_out, int out_size, void* d_ws, size_t ws_size,
                              hipStream_t stream) {
  const float* x       = (const float*)d_in[0];
  const float* Wqkv    = (const float*)d_in[1];
  const float* Wproj   = (const float*)d_in[2];
  const float* Wrouter = (const float*)d_in[3];
  const float* W1      = (const float*)d_in[4];
  const float* W2      = (const float*)d_in[5];
  const float* ln1_g   = (const float*)d_in[6];
  const float* ln1_b   = (const float*)d_in[7];
  const float* ln2_g   = (const float*)d_in[8];
  const float* ln2_b   = (const float*)d_in[9];
  float* out = (float*)d_out;

  char* p = (char*)d_ws;
  size_t off = 0;
  auto take = [&](size_t bytes) { void* r = p + off; off = alignup(off + bytes); return r; };

  h16*   WqkvT  = (h16*)take((size_t)3 * DIM * DIM * sizeof(h16));        // 2304x768
  h16*   WprojT = (h16*)take((size_t)DIM * DIM * sizeof(h16));
  h16*   W1T    = (h16*)take((size_t)NE * FF * DIM * sizeof(h16));        // e: 3072x768
  h16*   W2T    = (h16*)take((size_t)NE * DIM * FF * sizeof(h16));        // e: 768x3072
  h16*   xln1h  = (h16*)take((size_t)NTOK * DIM * sizeof(h16));
  h16*   qh     = (h16*)take((size_t)NTOK * DIM * sizeof(h16));
  h16*   kh     = (h16*)take((size_t)NTOK * DIM * sizeof(h16));
  h16*   vTh    = (h16*)take((size_t)NTOK * DIM * sizeof(h16));
  h16*   attnh  = (h16*)take((size_t)NTOK * DIM * sizeof(h16));
  float* xln2f  = (float*)take((size_t)NTOK * DIM * sizeof(float));
  h16*   xln2h  = (h16*)take((size_t)NTOK * DIM * sizeof(h16));
  float* logits = (float*)take((size_t)NTOK * NE * sizeof(float));
  int*   cnt    = (int*)take(NE * sizeof(int));
  int*   list   = (int*)take((size_t)NE * NTOK * sizeof(int));
  float* wgt    = (float*)take((size_t)NE * NTOK * sizeof(float));
  int*   invmap = (int*)take((size_t)NE * NTOK * sizeof(int));
  h16*   Ag     = (h16*)take((size_t)NTOK * DIM * sizeof(h16));
  h16*   Hbuf   = (h16*)take((size_t)NTOK * FF * sizeof(h16));
  float* ybuf   = (float*)take((size_t)NTOK * 2 * DIM * sizeof(float));
  (void)ws_size; (void)n_in; (void)in_sizes; (void)out_size;

  // 1) weights -> f16 transposed
  cvtT_kernel<<<dim3((DIM * 3 * DIM + 255) / 256, 1, 1), 256, 0, stream>>>(
      Wqkv, WqkvT, DIM, 3 * DIM, 0, 0);
  cvtT_kernel<<<dim3((DIM * DIM + 255) / 256, 1, 1), 256, 0, stream>>>(
      Wproj, WprojT, DIM, DIM, 0, 0);
  cvtT_kernel<<<dim3((DIM * FF + 255) / 256, 1, NE), 256, 0, stream>>>(
      W1, W1T, DIM, FF, (long)DIM * FF, (long)DIM * FF);
  cvtT_kernel<<<dim3((DIM * FF + 255) / 256, 1, NE), 256, 0, stream>>>(
      W2, W2T, FF, DIM, (long)DIM * FF, (long)DIM * FF);

  // 2) LN1 + attention path
  ln_kernel<<<NTOK, 256, 0, stream>>>(x, ln1_g, ln1_b, nullptr, xln1h);
  gemm_qkv<<<dim3(NTOK / 128, (3 * DIM) / 64), 128, 0, stream>>>(xln1h, WqkvT, qh, kh, vTh);
  attn_kernel<<<dim3(4 * NH, (TSEQ / 16) / 4), 128, 0, stream>>>(qh, kh, vTh, attnh);
  gemm_proj<<<dim3(NTOK / 128, DIM / 64), 128, 0, stream>>>(attnh, WprojT, x, out);

  // 3) LN2 + routing
  ln_kernel<<<NTOK, 256, 0, stream>>>(out, ln2_g, ln2_b, xln2f, xln2h);
  router_kernel<<<(NTOK * NE + 255) / 256, 256, 0, stream>>>(xln2f, Wrouter, logits);
  zero_cnt<<<1, 32, 0, stream>>>(cnt);
  topk_kernel<<<NTOK / 256, 256, 0, stream>>>(logits, cnt, list, wgt, invmap);

  // 4) MoE: only selected experts, sequential per expert
  for (int e = 0; e < NE; ++e) {
    gather_kernel<<<NTOK, 192, 0, stream>>>(xln2h, list + e * NTOK, cnt + e, Ag);
    ffn1_kernel<<<dim3(NTOK / 128, FF / 64), 128, 0, stream>>>(
        Ag, W1T + (size_t)e * FF * DIM, cnt + e, Hbuf);
    ffn2_kernel<<<dim3(NTOK / 128, DIM / 64), 128, 0, stream>>>(
        Hbuf, W2T + (size_t)e * DIM * FF, cnt + e, wgt + e * NTOK, invmap + e * NTOK, ybuf);
  }

  // 5) deterministic fixed-order residual sum
  final_kernel<<<(NTOK * DIM + 255) / 256, 256, 0, stream>>>(ybuf, out);
}
